// BiLSTM_Chars_CRF_30348238914063
// MI455X (gfx1250) — matrix operand order repaired and checked by reading, compiled, not run on hardware
//
#include <hip/hip_runtime.h>
#include <hip/hip_bf16.h>

// ---------------- problem constants ----------------
constexpr int kB    = 256;    // batch
constexpr int kT    = 512;    // time steps
constexpr int kEMB  = 64;
constexpr int kCHAR = 50;
constexpr int kDIN  = 114;    // EMB + CHAR
constexpr int kDPAD = 128;    // padded input dim (K for pre-GEMM)
constexpr int kRNN  = 128;
constexpr int kG4   = 512;    // 4*RNN gate width
constexpr int kTAGS = 9;
constexpr long long kMT = (long long)kB * kT;   // 131072 rows of x

// ---------------- workspace layout (bytes) ----------------
constexpr size_t OFF_X  = 0;
constexpr size_t SZ_X   = (size_t)kMT * kDPAD * 2;          // x bf16        33.6 MB
constexpr size_t OFF_WX = OFF_X + SZ_X;
constexpr size_t SZ_WX  = (size_t)2 * kDPAD * kG4 * 2;      // Wx bf16 blocked 256 KB
constexpr size_t OFF_WH = OFF_WX + SZ_WX;
constexpr size_t SZ_WH  = (size_t)2 * kRNN * kG4 * 2;       // Wh bf16 blocked 256 KB
constexpr size_t OFF_Z  = OFF_WH + SZ_WH;
constexpr size_t SZ_Z   = (size_t)2 * kMT * kG4 * 4;        // pre-acts f32  537 MB
constexpr size_t OFF_H  = OFF_Z + SZ_Z;
constexpr size_t SZ_H   = (size_t)2 * kT * kB * kRNN * 4;   // h_f/h_b f32   134 MB
constexpr size_t OFF_LG = OFF_H + SZ_H;
constexpr size_t SZ_LG  = (size_t)kMT * kTAGS * 4;          // logits f32    4.7 MB

// ---------------- WMMA types ----------------
typedef __bf16 v16bf __attribute__((ext_vector_type(16)));
typedef __bf16 v8bf  __attribute__((ext_vector_type(8)));
typedef float  v8f   __attribute__((ext_vector_type(8)));

#define CAT16(lo, hi) __builtin_shufflevector(lo, hi, 0,1,2,3,4,5,6,7,8,9,10,11,12,13,14,15)

// A-tile loader: 16x32 bf16 from row-major source (stride ldk elems).
// ISA 7.12.2: lanes 0-15 -> M=lane, elems {K0..7,K16..23}; lanes 16-31 -> {K8..15,K24..31}
// => per lane: two contiguous 16-byte loads.
__device__ __forceinline__ v16bf load_a_tile(const __bf16* base, int ldk, int lane, int k0) {
  const __bf16* rp = base + (size_t)(lane & 15) * ldk + k0 + ((lane >> 4) & 1) * 8;
  v8bf lo = *(const v8bf*)rp;
  v8bf hi = *(const v8bf*)(rp + 16);
  return CAT16(lo, hi);
}

// B-tile loader from pre-swizzled blocked storage: [nt][kk][lane 0..31][e 0..15],
// where element e of lane L holds W[k0 + (L>=16?8:0) + (e<8 ? e : 8+e)][nt*16 + L%16].
// => per lane: two contiguous 16-byte loads.
__device__ __forceinline__ v16bf load_b_blk(const __bf16* blk, int nt, int kk, int lane) {
  const __bf16* bp = blk + ((((size_t)nt * 4 + kk) * 32 + lane) << 4);
  v8bf lo = *(const v8bf*)bp;
  v8bf hi = *(const v8bf*)(bp + 8);
  return CAT16(lo, hi);
}

__device__ __forceinline__ float sigf(float x) { return 1.0f / (1.0f + __expf(-x)); }

// ---------------- kernel 1: pack weights -> bf16 B-operand blocked layout ----------------
// wx padded D_IN 114 -> 128 (zeros); both arrays become [dir][nt][kk][lane][e].
__global__ void pack_w_kernel(const float* __restrict__ Wx_f, const float* __restrict__ Wx_b,
                              const float* __restrict__ Wh_f, const float* __restrict__ Wh_b,
                              __bf16* __restrict__ wx, __bf16* __restrict__ wh) {
  const int per = kDPAD * kG4;                       // 65536 elems per dir
  int i = blockIdx.x * blockDim.x + threadIdx.x;     // 0 .. 4*per-1
  int arr = i / (2 * per);                           // 0 = wx, 1 = wh
  int rem = i % (2 * per);
  int dir = rem / per;
  int off = rem % per;
  int e    = off & 15;
  int lane = (off >> 4) & 31;
  int kk   = (off >> 9) & 3;
  int nt   = off >> 11;
  int col = nt * 16 + (lane & 15);
  int k   = kk * 32 + ((lane >> 4) & 1) * 8 + (e < 8 ? e : 8 + e);
  if (arr == 0) {
    const float* W = dir ? Wx_b : Wx_f;
    float v = (k < kDIN) ? W[(size_t)k * kG4 + col] : 0.0f;
    wx[(size_t)dir * per + off] = (__bf16)v;
  } else {
    const float* W = dir ? Wh_b : Wh_f;
    wh[(size_t)dir * per + off] = (__bf16)W[(size_t)k * kG4 + col];
  }
}

// ---------------- kernel 2: embedding gather + concat + pad -> x bf16 [MT][128] ----------------
__global__ void embed_kernel(const int* __restrict__ inputs, const float* __restrict__ chem,
                             const float* __restrict__ E, __bf16* __restrict__ x) {
  size_t i = (size_t)blockIdx.x * blockDim.x + threadIdx.x;   // MT*128 threads
  int col = (int)(i & (kDPAD - 1));
  size_t m = i >> 7;
  float v;
  if (col < kEMB)      v = E[(size_t)inputs[m] * kEMB + col];
  else if (col < kDIN) v = chem[m * kCHAR + (col - kEMB)];
  else                 v = 0.0f;
  x[i] = (__bf16)v;
}

// ---------------- kernel 3: pre-activation GEMM  Z[dir] = x @ Wx[dir] + b[dir] ----------------
// One 16x16 output tile per wave, K=128 -> 4 x v_wmma_f32_16x16x32_bf16.
__global__ void __launch_bounds__(256) pregemm_kernel(const __bf16* __restrict__ x,
                                                      const __bf16* __restrict__ wx,
                                                      const float* __restrict__ b_f,
                                                      const float* __restrict__ b_b,
                                                      float* __restrict__ Z) {
  const int lane = threadIdx.x & 31;
  const long long wave = ((long long)blockIdx.x * blockDim.x + threadIdx.x) >> 5;
  const int tilesPerDir = (int)(kMT / 16) * (kG4 / 16);       // 8192 * 32
  const int dir = (int)(wave / tilesPerDir);
  const int rem = (int)(wave % tilesPerDir);
  const int mt = rem >> 5;
  const int nt = rem & 31;
  const int m0 = mt * 16, n0 = nt * 16;

  const __bf16* A    = x + (size_t)m0 * kDPAD;
  const __bf16* Bblk = wx + (size_t)dir * kDPAD * kG4;
  const float* bias = dir ? b_b : b_f;
  const float bv = bias[n0 + (lane & 15)];

  v8f acc;
#pragma unroll
  for (int r = 0; r < 8; ++r) acc[r] = bv;

#pragma unroll
  for (int kk = 0; kk < 4; ++kk) {
    v16bf a = load_a_tile(A, kDPAD, lane, kk * 32);
    v16bf b = load_b_blk(Bblk, nt, kk, lane);
    acc = __builtin_amdgcn_wmma_f32_16x16x32_bf16(false, a, false, b, (short)0, acc, false, false);
  }

  float* Zt = Z + (size_t)dir * kMT * kG4;
  const int hi8 = ((lane >> 4) & 1) * 8;
#pragma unroll
  for (int r = 0; r < 8; ++r)
    Zt[(size_t)(m0 + r + hi8) * kG4 + n0 + (lane & 15)] = acc[r];
}

// ---------------- kernel 4: persistent LSTM recurrence (one block per direction) ----------------
// LDS: Wh bf16 blocked (128KB, filled by async global->LDS DMA) + h bf16 [256][128] (64KB).
// 32 waves; each wave owns 4 tasks (batch-tile, hidden-tile); 16 WMMAs/task/step; c in VGPRs.
__global__ void __launch_bounds__(1024) lstm_kernel(const float* __restrict__ Z,
                                                    const __bf16* __restrict__ wh,
                                                    float* __restrict__ hbuf) {
  extern __shared__ char smem[];
  __bf16* sWh = (__bf16*)smem;                              // blocked, 128*512
  __bf16* sh  = (__bf16*)(smem + (size_t)kRNN * kG4 * 2);   // 256*128

  const int dir  = blockIdx.x;
  const int tid  = threadIdx.x;
  const int wave = tid >> 5;
  const int lane = tid & 31;
  const int hi8  = ((lane >> 4) & 1) * 8;

  // Async DMA Wh (blocked) into LDS: each lane moves 16B straight to LDS (no VGPR data),
  // tracked by ASYNCcnt (CDNA5 ISA).
  {
    const __bf16* whp = wh + (size_t)dir * kRNN * kG4;
    for (int i = tid * 8; i < kRNN * kG4; i += blockDim.x * 8) {
      unsigned lds_off = (unsigned)(uintptr_t)(sWh + i);
      unsigned long long ga = (unsigned long long)(uintptr_t)(whp + i);
      asm volatile("global_load_async_to_lds_b128 %0, %1, off"
                   :: "v"(lds_off), "v"(ga) : "memory");
    }
    asm volatile("s_wait_asynccnt 0x0" ::: "memory");
  }
  for (int i = tid; i < kB * kRNN; i += blockDim.x) sh[i] = (__bf16)0.0f;
  __syncthreads();

  const float* Zd = Z + (size_t)dir * kMT * kG4;
  float* hd = hbuf + (size_t)dir * kT * kB * kRNN;

  v8f c[4];
#pragma unroll
  for (int q = 0; q < 4; ++q)
#pragma unroll
    for (int r = 0; r < 8; ++r) c[q][r] = 0.0f;

  for (int t = 0; t < kT; ++t) {
    const int tf = dir ? (kT - 1 - t) : t;
    float hval[4][8];

#pragma unroll
    for (int qi = 0; qi < 4; ++qi) {
      const int q  = wave * 4 + qi;
      const int mt = q >> 3;            // 16 batch tiles
      const int jt = q & 7;             // 8 hidden tiles

      v8f acc[4];
      // seed accumulators with precomputed x@Wx + b
#pragma unroll
      for (int g = 0; g < 4; ++g) {
        const int n0 = g * kRNN + jt * 16;
#pragma unroll
        for (int r = 0; r < 8; ++r) {
          const int brow = mt * 16 + r + hi8;
          acc[g][r] = Zd[((size_t)brow * kT + tf) * kG4 + n0 + (lane & 15)];
        }
      }
      // prefetch next step's Z rows while WMMAs run
      if (t + 1 < kT) {
        const int tn = dir ? (tf - 1) : (tf + 1);
        __builtin_prefetch(&Zd[((size_t)(mt * 16 + hi8) * kT + tn) * kG4 + jt * 16], 0, 1);
      }
      // h_{t-1} @ Wh : K = 128 -> 4 chunks of 32
#pragma unroll
      for (int kk = 0; kk < 4; ++kk) {
        v16bf a = load_a_tile(sh + (size_t)mt * 16 * kRNN, kRNN, lane, kk * 32);
#pragma unroll
        for (int g = 0; g < 4; ++g) {
          v16bf b = load_b_blk(sWh, g * 8 + jt, kk, lane);
          acc[g] = __builtin_amdgcn_wmma_f32_16x16x32_bf16(false, a, false, b, (short)0,
                                                           acc[g], false, false);
        }
      }
      // LSTM cell update; c stays in registers
#pragma unroll
      for (int r = 0; r < 8; ++r) {
        const float zi = acc[0][r], zf = acc[1][r], zg = acc[2][r], zo = acc[3][r];
        const float cn = sigf(zf) * c[qi][r] + sigf(zi) * tanhf(zg);
        c[qi][r] = cn;
        hval[qi][r] = sigf(zo) * tanhf(cn);
      }
    }

    __syncthreads();   // all reads of h_{t-1} from LDS complete

#pragma unroll
    for (int qi = 0; qi < 4; ++qi) {
      const int q  = wave * 4 + qi;
      const int mt = q >> 3;
      const int jt = q & 7;
#pragma unroll
      for (int r = 0; r < 8; ++r) {
        const int brow = mt * 16 + r + hi8;
        const int col  = jt * 16 + (lane & 15);
        sh[(size_t)brow * kRNN + col] = (__bf16)hval[qi][r];
        hd[((size_t)tf * kB + brow) * kRNN + col] = hval[qi][r];
      }
    }
    __syncthreads();   // h_t visible to all waves
  }
}

// ---------------- kernel 5: logits = [h_f;h_b] @ Wd + bd, fully parallel ----------------
__global__ void logits_kernel(const float* __restrict__ hbuf, const float* __restrict__ Wd,
                              const float* __restrict__ bd, float* __restrict__ lg) {
  size_t i = (size_t)blockIdx.x * blockDim.x + threadIdx.x;   // MT*TAGS threads
  if (i >= (size_t)kMT * kTAGS) return;
  int j = (int)(i % kTAGS);
  size_t m = i / kTAGS;                 // m = b*T + t
  int b = (int)(m / kT), t = (int)(m % kT);
  const float* hfp = hbuf + ((size_t)t * kB + b) * kRNN;
  const float* hbp = hbuf + (size_t)kT * kB * kRNN + ((size_t)t * kB + b) * kRNN;
  float s = bd[j];
  for (int k = 0; k < kRNN; ++k) s += hfp[k] * Wd[(size_t)k * kTAGS + j];
  for (int k = 0; k < kRNN; ++k) s += hbp[k] * Wd[(size_t)(kRNN + k) * kTAGS + j];
  lg[m * kTAGS + j] = s;
}

// ---------------- kernel 6: CRF log-likelihood scan (one thread per sequence) ----------------
__global__ void crf_kernel(const float* __restrict__ lgbuf, const int* __restrict__ labels,
                           const float* __restrict__ trans, float* __restrict__ out) {
  const int b = blockIdx.x * blockDim.x + threadIdx.x;
  if (b >= kB) return;

  int len = 0;
  for (int t = 0; t < kT; ++t) len += (labels[(size_t)b * kT + t] != 0);

  float unary = 0.0f, binary = 0.0f;
  float alpha[kTAGS];
  int prev_lab = 0;

  for (int t = 0; t < kT; ++t) {
    const float* lg = lgbuf + ((size_t)b * kT + t) * kTAGS;
    const int lab = labels[(size_t)b * kT + t];
    const float m = (t < len) ? 1.0f : 0.0f;
    unary += m * lg[lab];
    if (t >= 1) binary += m * trans[prev_lab * kTAGS + lab];
    prev_lab = lab;

    if (t == 0) {
#pragma unroll
      for (int j = 0; j < kTAGS; ++j) alpha[j] = lg[j];
    } else if (t < len) {
      float na[kTAGS];
#pragma unroll
      for (int j = 0; j < kTAGS; ++j) {
        float mx = -1e30f;
#pragma unroll
        for (int i = 0; i < kTAGS; ++i) mx = fmaxf(mx, alpha[i] + trans[i * kTAGS + j]);
        float s = 0.0f;
#pragma unroll
        for (int i = 0; i < kTAGS; ++i) s += __expf(alpha[i] + trans[i * kTAGS + j] - mx);
        na[j] = mx + __logf(s) + lg[j];
      }
#pragma unroll
      for (int j = 0; j < kTAGS; ++j) alpha[j] = na[j];
    }
  }
  float mx = -1e30f;
#pragma unroll
  for (int j = 0; j < kTAGS; ++j) mx = fmaxf(mx, alpha[j]);
  float s = 0.0f;
#pragma unroll
  for (int j = 0; j < kTAGS; ++j) s += __expf(alpha[j] - mx);
  const float lognorm = (len > 0) ? (mx + __logf(s)) : 0.0f;
  out[b] = unary + binary - lognorm;
  if (b < kTAGS * kTAGS) out[kB + b] = trans[b];   // second tuple output
}

// ---------------- host launcher ----------------
extern "C" void kernel_launch(void* const* d_in, const int* in_sizes, int n_in,
                              void* d_out, int out_size, void* d_ws, size_t ws_size,
                              hipStream_t stream) {
  const int*   inputs = (const int*)  d_in[0];
  const float* chem   = (const float*)d_in[1];
  const int*   labels = (const int*)  d_in[2];
  const float* E      = (const float*)d_in[3];
  const float* Wx_f   = (const float*)d_in[4];
  const float* Wh_f   = (const float*)d_in[5];
  const float* b_f    = (const float*)d_in[6];
  const float* Wx_b   = (const float*)d_in[7];
  const float* Wh_b   = (const float*)d_in[8];
  const float* b_b    = (const float*)d_in[9];
  const float* Wd     = (const float*)d_in[10];
  const float* bd     = (const float*)d_in[11];
  const float* trans  = (const float*)d_in[12];

  char* ws = (char*)d_ws;
  __bf16* x    = (__bf16*)(ws + OFF_X);
  __bf16* wx   = (__bf16*)(ws + OFF_WX);
  __bf16* wh   = (__bf16*)(ws + OFF_WH);
  float*  Z    = (float*)(ws + OFF_Z);
  float*  hbuf = (float*)(ws + OFF_H);
  float*  lg   = (float*)(ws + OFF_LG);
  float*  out  = (float*)d_out;

  // 1) weights -> bf16 blocked B-operand layout
  pack_w_kernel<<<(2 * 2 * kDPAD * kG4) / 256, 256, 0, stream>>>(Wx_f, Wx_b, Wh_f, Wh_b, wx, wh);

  // 2) embedding + concat + pad
  embed_kernel<<<(unsigned)((kMT * kDPAD) / 256), 256, 0, stream>>>(inputs, chem, E, x);

  // 3) pre-activation GEMM: 2 dirs * 8192 * 32 tiles, 8 waves/block
  {
    long long tiles = 2LL * (kMT / 16) * (kG4 / 16);
    pregemm_kernel<<<(unsigned)(tiles / 8), 256, 0, stream>>>(x, wx, b_f, b_b, Z);
  }

  // 4) persistent recurrence: one block per direction, 192KB LDS
  {
    size_t smem = (size_t)kRNN * kG4 * 2 + (size_t)kB * kRNN * 2;  // 196608 B
    lstm_kernel<<<2, 1024, smem, stream>>>(Z, wh, hbuf);
  }

  // 5) logits (parallel) then 6) CRF scan
  {
    long long n = kMT * kTAGS;
    logits_kernel<<<(unsigned)((n + 255) / 256), 256, 0, stream>>>(hbuf, Wd, bd, lg);
  }
  crf_kernel<<<8, 32, 0, stream>>>(lg, labels, trans, out);
}
